// Mamba2SSDTemporalCore_13494787244125
// MI455X (gfx1250) — compile-verified, matrix operand order checked by reading, not run on hardware
//
#include <hip/hip_runtime.h>
#include <math.h>
#include <stddef.h>

// ---------------------------------------------------------------------------
// Mamba-2 SSD block, CDNA5 (gfx1250, wave32) WMMA implementation.
// All matmuls -> v_wmma_f32_16x16x32_f16 (f32 accumulate).
// Tile loads in the SSD kernels use GLOBAL_LOAD_ASYNC_TO_LDS_B128 (ASYNCcnt).
// ---------------------------------------------------------------------------

typedef __attribute__((ext_vector_type(16))) _Float16 v16h;
typedef __attribute__((ext_vector_type(8)))  _Float16 v8h;
typedef __attribute__((ext_vector_type(8)))  float    v8f;

#define LDZX 2304   // padded leading dim of zxbcdt (2192 -> 18*128)

// Build an A/B fragment when the 16 K-elements this lane needs are two
// contiguous runs of 8 halves: [k0 .. k0+7] and [k0+16 .. k0+23]
// (k0 already includes the lane's khalf*8). Matches the 16-bit 16x32 layout.
__device__ __forceinline__ v16h frag_rows(const _Float16* p) {
  v8h lo = *(const v8h*)(p);
  v8h hi = *(const v8h*)(p + 16);
  v16h r;
#pragma unroll
  for (int i = 0; i < 8; ++i) { r[i] = lo[i]; r[8 + i] = hi[i]; }
  return r;
}

// Same fragment but gathered with a row stride (for transposed operands in LDS).
__device__ __forceinline__ v16h frag_gather(const _Float16* p, int stride) {
  v16h r;
#pragma unroll
  for (int i = 0; i < 8; ++i) { r[i] = p[i * stride]; r[8 + i] = p[(16 + i) * stride]; }
  return r;
}

__device__ __forceinline__ v8f wmma_f16(v16h a, v16h b, v8f c) {
  return __builtin_amdgcn_wmma_f32_16x16x32_f16(false, a, false, b, (short)0, c,
                                                false, false);
}

// --- CDNA5 async global->LDS copy (tracked by ASYNCcnt) --------------------
__device__ __forceinline__ unsigned to_lds_off(const void* p) {
  return (unsigned)(unsigned long long)(__attribute__((address_space(3))) const void*)p;
}

__device__ __forceinline__ void async_copy_b128(unsigned lds_off, const void* gptr) {
  asm volatile("global_load_async_to_lds_b128 %0, %1, off"
               :: "v"(lds_off), "v"((unsigned long long)gptr)
               : "memory");
}

__device__ __forceinline__ void wait_async() {
  asm volatile("s_wait_asynccnt 0x0" ::: "memory");
}

// ---------------------------------------------------------------------------
// Generic C = A * B^T GEMM.  A: MxK row-major f16, B: NxK row-major f16,
// C: MxN row-major f32.  Block = 128(M) x 128(N), 8 waves (256 thr),
// wave tile 32x64, K step 32.  M % 128 == 0, N % 128 == 0, K % 32 == 0.
// ---------------------------------------------------------------------------
__global__ void __launch_bounds__(256)
gemm_f16nt(const _Float16* __restrict__ A, const _Float16* __restrict__ B,
           float* __restrict__ C, int M, int N, int K) {
  const int lane = threadIdx.x & 31;
  const int wv   = threadIdx.x >> 5;
  const int wm = wv & 3, wn = wv >> 2;               // 4 x 2 wave grid
  const int m0 = blockIdx.x * 128 + wm * 32;
  const int n0 = blockIdx.y * 128 + wn * 64;
  const int l16 = lane & 15, khalf = lane >> 4;

  v8f acc[2][4] = {};
  for (int kk = 0; kk < K; kk += 32) {
    const int k0 = kk + khalf * 8;
    if (kk + 32 < K) {
      __builtin_prefetch(A + (size_t)(m0 + l16) * K + kk + 32, 0, 1);
      __builtin_prefetch(B + (size_t)(n0 + l16) * K + kk + 32, 0, 1);
    }
    v16h af[2], bf[4];
#pragma unroll
    for (int mt = 0; mt < 2; ++mt)
      af[mt] = frag_rows(A + (size_t)(m0 + mt * 16 + l16) * K + k0);
#pragma unroll
    for (int nt = 0; nt < 4; ++nt)
      bf[nt] = frag_rows(B + (size_t)(n0 + nt * 16 + l16) * K + k0);
#pragma unroll
    for (int mt = 0; mt < 2; ++mt)
#pragma unroll
      for (int nt = 0; nt < 4; ++nt)
        acc[mt][nt] = wmma_f16(af[mt], bf[nt], acc[mt][nt]);
  }
  // Dims are padded by the caller: no bounds checks -> branch-free epilogue.
#pragma unroll
  for (int mt = 0; mt < 2; ++mt) {
#pragma unroll
    for (int nt = 0; nt < 4; ++nt) {
      const int n  = n0 + nt * 16 + l16;
      const int mb = m0 + mt * 16 + khalf * 8;
      float* Cp = C + (size_t)mb * N + n;
#pragma unroll
      for (int r = 0; r < 8; ++r) Cp[(size_t)r * N] = acc[mt][nt][r];
    }
  }
}

// ---------------------------------------------------------------------------
// f32 -> f16 converts
// ---------------------------------------------------------------------------
__global__ void cvt16(const float* __restrict__ s, _Float16* __restrict__ d, int n) {
  int i = blockIdx.x * 256 + threadIdx.x;
  if (i < n) d[i] = (_Float16)s[i];
}

__global__ void cvt16_pad_rows(const float* __restrict__ s, _Float16* __restrict__ d,
                               int rows, int rows_pad, int cols) {
  int i = blockIdx.x * 256 + threadIdx.x;
  if (i >= rows_pad * cols) return;
  int r = i / cols;
  d[i] = (r < rows) ? (_Float16)s[i] : (_Float16)0.f;
}

// ---------------------------------------------------------------------------
// dt = softplus(dt_raw + dt_bias)
// ---------------------------------------------------------------------------
__global__ void dt_softplus(const float* __restrict__ zx,
                            const float* __restrict__ dt_bias,
                            float* __restrict__ dt) {
  int idx = blockIdx.x * 256 + threadIdx.x;
  if (idx >= 32768 * 16) return;
  int t = idx >> 4, h = idx & 15;
  float v = zx[(size_t)t * LDZX + 2176 + h] + dt_bias[h];
  dt[idx] = (v > 20.f) ? v : log1pf(__expf(v));
}

// ---------------------------------------------------------------------------
// Causal depthwise conv (K=4) + SiLU, split into x / B / C (f16 outputs).
// ---------------------------------------------------------------------------
__global__ void conv_silu(const float* __restrict__ zx,
                          const float* __restrict__ cw,
                          const float* __restrict__ cb,
                          _Float16* __restrict__ xc,
                          _Float16* __restrict__ Bc,
                          _Float16* __restrict__ Cc) {
  int idx = blockIdx.x * 256 + threadIdx.x;
  if (idx >= 32768 * 1152) return;
  int t = idx / 1152, ch = idx % 1152;
  int tloc = t & 8191;                                   // position in sequence
  const float* col = zx + (size_t)t * LDZX + 1024 + ch;
  float acc = cb[ch];
#pragma unroll
  for (int k = 0; k < 4; ++k) {
    int d = k - 3;
    if (tloc + d >= 0) acc += col[(ptrdiff_t)d * LDZX] * cw[ch * 4 + k];
  }
  float s = acc / (1.f + __expf(-acc));
  _Float16 hv = (_Float16)s;
  if (ch < 1024)      xc[(size_t)t * 1024 + ch] = hv;
  else if (ch < 1088) Bc[(size_t)t * 64 + (ch - 1024)] = hv;
  else                Cc[(size_t)t * 64 + (ch - 1088)] = hv;
}

// ---------------------------------------------------------------------------
// Intra-chunk SSD: one block per (chunk, head).  Three 64x64x64 WMMA matmuls
// chained entirely through LDS:
//   G = (C B^T) o exp(segsum(a))          -> sG
//   Y_diag = G X                          -> Y  (global)
//   S = B^T diag(exp(cum_last-cum)) X     -> states (global)
// B/C tiles (8 KB contiguous each) arrive via async global->LDS DMA.
// ---------------------------------------------------------------------------
__global__ void __launch_bounds__(128)
ssd_intra(const _Float16* __restrict__ xconv, const _Float16* __restrict__ Bc,
          const _Float16* __restrict__ Cc, const float* __restrict__ dtb,
          const float* __restrict__ A_log, float* __restrict__ Y,
          float* __restrict__ states, float* __restrict__ cumA) {
  __shared__ __align__(16) _Float16 sB[64 * 64];
  __shared__ __align__(16) _Float16 sC[64 * 64];       // later reused as Xw
  __shared__ __align__(16) _Float16 sX[64 * 64];
  __shared__ __align__(16) _Float16 sG[64 * 64];
  __shared__ float cum[64], dts[64];

  const int cc = blockIdx.x;            // global chunk index (b*128 + c)
  const int h  = blockIdx.y;
  const int t0 = cc * 64;
  const int tid  = threadIdx.x;
  const int lane = tid & 31, wv = tid >> 5;
  const int l16 = lane & 15, khalf = lane >> 4;

  // Kick off async DMA of the chunk's B and C tiles into LDS.
  {
    const unsigned sB0 = to_lds_off(sB);
    const unsigned sC0 = to_lds_off(sC);
    const char* gB = (const char*)(Bc + (size_t)t0 * 64);
    const char* gC = (const char*)(Cc + (size_t)t0 * 64);
#pragma unroll
    for (int rep = 0; rep < 4; ++rep) {
      const int off = (tid + rep * 128) * 16;
      async_copy_b128(sB0 + off, gB + off);
      async_copy_b128(sC0 + off, gC + off);
    }
  }

  if (tid < 64) dts[tid] = dtb[(size_t)(t0 + tid) * 16 + h];
  __syncthreads();
  if (tid == 0) {
    float a = -__expf(A_log[h]);
    float s = 0.f;
    for (int t = 0; t < 64; ++t) { s += a * dts[t]; cum[t] = s; }
  }
  __syncthreads();
  if (tid < 64) cumA[((size_t)cc * 16 + h) * 64 + tid] = cum[tid];

  for (int i = tid; i < 64 * 64; i += 128) {
    int s = i >> 6, j = i & 63;
    float xv = (float)xconv[(size_t)(t0 + s) * 1024 + h * 64 + j];
    sX[i] = (_Float16)(xv * dts[s]);                    // X = x * dt
  }
  wait_async();
  __syncthreads();

  // ---- G = C * B^T, masked by decay ----
  {
    v8f acc[4] = {};
#pragma unroll
    for (int ks = 0; ks < 2; ++ks) {
      const int k0 = ks * 32 + khalf * 8;
      v16h a = frag_rows(&sC[(16 * wv + l16) * 64 + k0]);
#pragma unroll
      for (int nt = 0; nt < 4; ++nt) {
        v16h b = frag_rows(&sB[(16 * nt + l16) * 64 + k0]);
        acc[nt] = wmma_f16(a, b, acc[nt]);
      }
    }
#pragma unroll
    for (int nt = 0; nt < 4; ++nt) {
      const int scol = 16 * nt + l16;
#pragma unroll
      for (int r = 0; r < 8; ++r) {
        const int l = 16 * wv + khalf * 8 + r;
        float v = (scol <= l) ? acc[nt][r] * __expf(cum[l] - cum[scol]) : 0.f;
        sG[l * 64 + scol] = (_Float16)v;
      }
    }
  }
  __syncthreads();

  // ---- Y_diag = G @ X; also build Xw = X * exp(cum_last - cum_s) into sC ----
  {
    const float clast = cum[63];
    for (int i = tid; i < 64 * 64; i += 128) {
      int s = i >> 6;
      sC[i] = (_Float16)((float)sX[i] * __expf(clast - cum[s]));
    }
    v8f acc[4] = {};
#pragma unroll
    for (int ks = 0; ks < 2; ++ks) {
      const int k0 = ks * 32 + khalf * 8;
      v16h a = frag_rows(&sG[(16 * wv + l16) * 64 + k0]);
#pragma unroll
      for (int nt = 0; nt < 4; ++nt) {
        v16h b = frag_gather(&sX[k0 * 64 + 16 * nt + l16], 64);
        acc[nt] = wmma_f16(a, b, acc[nt]);
      }
    }
#pragma unroll
    for (int nt = 0; nt < 4; ++nt) {
      const int p = 16 * nt + l16;
#pragma unroll
      for (int r = 0; r < 8; ++r) {
        const int l = 16 * wv + khalf * 8 + r;
        Y[(size_t)(t0 + l) * 1024 + h * 64 + p] = acc[nt][r];
      }
    }
  }
  __syncthreads();

  // ---- S = B^T @ Xw ----
  {
    v8f acc[4] = {};
#pragma unroll
    for (int ks = 0; ks < 2; ++ks) {
      const int k0 = ks * 32 + khalf * 8;
      v16h a = frag_gather(&sB[k0 * 64 + 16 * wv + l16], 64);
#pragma unroll
      for (int nt = 0; nt < 4; ++nt) {
        v16h b = frag_gather(&sC[k0 * 64 + 16 * nt + l16], 64);  // sC == Xw
        acc[nt] = wmma_f16(a, b, acc[nt]);
      }
    }
    float* Sdst = states + ((size_t)cc * 16 + h) * 4096;
#pragma unroll
    for (int nt = 0; nt < 4; ++nt) {
      const int p = 16 * nt + l16;
#pragma unroll
      for (int r = 0; r < 8; ++r) {
        const int n = 16 * wv + khalf * 8 + r;
        Sdst[n * 64 + p] = acc[nt][r];
      }
    }
  }
}

// ---------------------------------------------------------------------------
// Inter-chunk SSD: one block per (batch, head).  The 64x64 running state stays
// in LDS for all 128 chunks.  Y_off = diag(exp(cum)) C @ S_prev  (WMMA), then
// Y_final = Y_diag + Y_off + D*x; state update S = exp(cum_last)*S + S_chunk.
// ---------------------------------------------------------------------------
__global__ void __launch_bounds__(128)
ssd_inter(const _Float16* __restrict__ xconv, const _Float16* __restrict__ Cc,
          const float* __restrict__ Dp, const float* __restrict__ states,
          const float* __restrict__ cumA, float* __restrict__ Y) {
  __shared__ __align__(16) float    Sst[64 * 64];
  __shared__ __align__(16) _Float16 S16[64 * 64];
  __shared__ __align__(16) _Float16 sC[64 * 64];
  __shared__ float cum[64];

  const int b = blockIdx.x >> 4, h = blockIdx.x & 15;
  const int tid  = threadIdx.x;
  const int lane = tid & 31, wv = tid >> 5;
  const int l16 = lane & 15, khalf = lane >> 4;
  const float Dh = Dp[h];
  const unsigned sC0 = to_lds_off(sC);

  for (int i = tid; i < 4096; i += 128) Sst[i] = 0.f;

  for (int c = 0; c < 128; ++c) {
    const int cc = b * 128 + c;
    const int t0 = cc * 64;
    __syncthreads();
    // async DMA of this chunk's C tile (8 KB contiguous) into LDS
    {
      const char* gC = (const char*)(Cc + (size_t)t0 * 64);
#pragma unroll
      for (int rep = 0; rep < 4; ++rep) {
        const int off = (tid + rep * 128) * 16;
        async_copy_b128(sC0 + off, gC + off);
      }
    }
    if (tid < 64) cum[tid] = cumA[((size_t)cc * 16 + h) * 64 + tid];
    for (int i = tid; i < 4096; i += 128) S16[i] = (_Float16)Sst[i];
    wait_async();
    __syncthreads();

    v8f acc[4] = {};
#pragma unroll
    for (int ks = 0; ks < 2; ++ks) {
      const int k0 = ks * 32 + khalf * 8;
      v16h a = frag_rows(&sC[(16 * wv + l16) * 64 + k0]);
#pragma unroll
      for (int nt = 0; nt < 4; ++nt) {
        v16h bfr = frag_gather(&S16[k0 * 64 + 16 * nt + l16], 64);
        acc[nt] = wmma_f16(a, bfr, acc[nt]);
      }
    }
#pragma unroll
    for (int nt = 0; nt < 4; ++nt) {
      const int p = 16 * nt + l16;
#pragma unroll
      for (int r = 0; r < 8; ++r) {
        const int t = 16 * wv + khalf * 8 + r;
        const size_t yi = (size_t)(t0 + t) * 1024 + h * 64 + p;
        const float xv = (float)xconv[yi];
        Y[yi] = Y[yi] + __expf(cum[t]) * acc[nt][r] + Dh * xv;
      }
    }
    __syncthreads();
    const float dec = __expf(cum[63]);
    const float* Sc = states + ((size_t)cc * 16 + h) * 4096;
    for (int i = tid; i < 4096; i += 128) Sst[i] = dec * Sst[i] + Sc[i];
  }
}

// ---------------------------------------------------------------------------
// Fused RMS-ish norm (||y||/sqrt(D)) + SiLU(z) gate; emits f16 for GEMM2.
// One block per token.
// ---------------------------------------------------------------------------
__global__ void __launch_bounds__(256)
gate_norm(const float* __restrict__ Y, const float* __restrict__ zx,
          const float* __restrict__ norm_w, _Float16* __restrict__ yg) {
  __shared__ float red[256];
  const int t = blockIdx.x;
  const int tid = threadIdx.x;
  float v[4];
  float ss = 0.f;
#pragma unroll
  for (int i = 0; i < 4; ++i) {
    v[i] = Y[(size_t)t * 1024 + tid + i * 256];
    ss += v[i] * v[i];
  }
  red[tid] = ss;
  __syncthreads();
  for (int off = 128; off > 0; off >>= 1) {
    if (tid < off) red[tid] += red[tid + off];
    __syncthreads();
  }
  const float rms = sqrtf(red[0]) * 0.03125f;   // * D_INNER^-0.5 (1/32)
  const float inv = 1.f / (rms + 1e-5f);
#pragma unroll
  for (int i = 0; i < 4; ++i) {
    const int ch = tid + i * 256;
    const float z = zx[(size_t)t * LDZX + ch];
    const float sil = z / (1.f + __expf(-z));
    yg[(size_t)t * 1024 + ch] = (_Float16)(norm_w[ch] * v[i] * inv * sil);
  }
}

// ---------------------------------------------------------------------------
extern "C" void kernel_launch(void* const* d_in, const int* in_sizes, int n_in,
                              void* d_out, int out_size, void* d_ws, size_t ws_size,
                              hipStream_t stream) {
  const float* z_seq   = (const float*)d_in[0];
  const float* W_in    = (const float*)d_in[1];
  const float* conv_w  = (const float*)d_in[2];
  const float* conv_b  = (const float*)d_in[3];
  const float* A_log   = (const float*)d_in[4];
  const float* dt_bias = (const float*)d_in[5];
  const float* Dp      = (const float*)d_in[6];
  const float* norm_w  = (const float*)d_in[7];
  const float* W_out   = (const float*)d_in[8];
  float* out = (float*)d_out;
  (void)in_sizes; (void)n_in; (void)out_size; (void)ws_size;

  char* ws = (char*)d_ws;
  size_t off = 0;
  auto alloc = [&](size_t bytes) -> void* {
    void* p = ws + off;
    off = (off + bytes + 255) & ~(size_t)255;
    return p;
  };
  _Float16* z16    = (_Float16*)alloc((size_t)32768 * 512 * 2);
  _Float16* Win16  = (_Float16*)alloc((size_t)LDZX * 512 * 2);   // N padded 2192->2304
  _Float16* Wout16 = (_Float16*)alloc((size_t)512 * 1024 * 2);
  float*    zx     = (float*)   alloc((size_t)32768 * LDZX * 4); // zxbcdt (padded ld)
  float*    dtb    = (float*)   alloc((size_t)32768 * 16 * 4);
  _Float16* xc     = (_Float16*)alloc((size_t)32768 * 1024 * 2);
  _Float16* Bc     = (_Float16*)alloc((size_t)32768 * 64 * 2);
  _Float16* Cc     = (_Float16*)alloc((size_t)32768 * 64 * 2);
  float*    cumA   = (float*)   alloc((size_t)512 * 16 * 64 * 4);
  float*    st     = (float*)   alloc((size_t)512 * 16 * 4096 * 4);
  float*    Ybuf   = (float*)   alloc((size_t)32768 * 1024 * 4);
  _Float16* yg     = (_Float16*)alloc((size_t)32768 * 1024 * 2);

  cvt16<<<(32768 * 512 + 255) / 256, 256, 0, stream>>>(z_seq, z16, 32768 * 512);
  cvt16_pad_rows<<<(LDZX * 512 + 255) / 256, 256, 0, stream>>>(W_in, Win16, 2192, LDZX, 512);
  cvt16<<<(512 * 1024 + 255) / 256, 256, 0, stream>>>(W_out, Wout16, 512 * 1024);

  gemm_f16nt<<<dim3(32768 / 128, LDZX / 128), 256, 0, stream>>>(z16, Win16, zx,
                                                                32768, LDZX, 512);
  dt_softplus<<<(32768 * 16 + 255) / 256, 256, 0, stream>>>(zx, dt_bias, dtb);
  conv_silu<<<(32768 * 1152 + 255) / 256, 256, 0, stream>>>(zx, conv_w, conv_b,
                                                            xc, Bc, Cc);
  ssd_intra<<<dim3(512, 16), 128, 0, stream>>>(xc, Bc, Cc, dtb, A_log, Ybuf, st, cumA);
  ssd_inter<<<64, 128, 0, stream>>>(xc, Cc, Dp, st, cumA, Ybuf);
  gate_norm<<<32768, 256, 0, stream>>>(Ybuf, zx, norm_w, yg);
  gemm_f16nt<<<dim3(32768 / 128, 512 / 128), 256, 0, stream>>>(yg, Wout16, out,
                                                               32768, 512, 1024);
}
